// Cos_Attn_57638461112506
// MI455X (gfx1250) — matrix operand order repaired and checked by reading, not compiled, not run on hardware
//
#include <hip/hip_runtime.h>
#include <hip/hip_bf16.h>

typedef __attribute__((ext_vector_type(16))) _Float16 v16h;
typedef __attribute__((ext_vector_type(8)))  _Float16 v8h;
typedef __attribute__((ext_vector_type(8)))  float    v8f;
typedef __attribute__((ext_vector_type(4)))  float    v4f;

#define BATCH 4
#define CDIM  64
#define NDIM  4096

// ---------------------------------------------------------------------------
// Phase 1: per column (b, n): f32 sum of squares -> rnorm = rsqrt(ss),
// and write an f16 transposed copy Xt[b][n][c] (c contiguous, 128B per row)
// so the WMMA fragment loads in phase 2 are contiguous 16B/32B per lane.
// ---------------------------------------------------------------------------
__global__ __launch_bounds__(256) void prep_kernel(const float* __restrict__ x,
                                                   _Float16* __restrict__ xt,
                                                   float* __restrict__ rnorm) {
    const int idx = blockIdx.x * 256 + threadIdx.x;   // idx = b*NDIM + n
    const int b = idx >> 12;
    const int n = idx & (NDIM - 1);

    const float* col = x + ((size_t)b << 18) + n;     // b*CDIM*NDIM, stride NDIM over c
    _Float16* row = xt + ((size_t)idx << 6);          // 64 halves per row

    float ss = 0.0f;
#pragma unroll
    for (int cc = 0; cc < 8; ++cc) {
        v8h buf;
#pragma unroll
        for (int k = 0; k < 8; ++k) {
            float v = col[(size_t)(cc * 8 + k) << 12];
            ss += v * v;
            buf[k] = (_Float16)v;
        }
        *(v8h*)(row + cc * 8) = buf;                  // 16B store, 16B aligned
    }
    rnorm[idx] = rsqrtf(ss);
}

// ---------------------------------------------------------------------------
// Phase 2: one wave computes a 16(i) x 64(j) strip of the output:
// A fragment (16x64 over two K=32 steps) loaded once, reused for 4 j-tiles.
//
// ISA fragment layouts (wave32, 16-bit A 16x32 / B 32x16 / f32 C 16x16):
//   A: lane (g = lane>>4, m = lane&15) holds M=m, K in [16g, 16g+16) (contiguous order used here)
//   B: lane (g, n=m) holds N=n, K in [16g, 16g+16) (same order as A; not the ISA table order)
//   C: VGPR r, lane (g, m) holds row (r + 8g), col m
// ---------------------------------------------------------------------------
__global__ __launch_bounds__(256) void gram_kernel(const _Float16* __restrict__ xt,
                                                   const float* __restrict__ rnorm,
                                                   float* __restrict__ out) {
    const int lane = threadIdx.x & 31;
    const int wave = threadIdx.x >> 5;
    const int g = lane >> 4;
    const int m = lane & 15;
    const int b  = blockIdx.z;
    const int i0 = blockIdx.y << 4;
    const int j0base = ((blockIdx.x << 3) + wave) << 6;   // 64 columns per wave

    const _Float16* __restrict__ xb = xt + ((size_t)b << 18);    // b*NDIM*CDIM
    const float* __restrict__ rb = rnorm + ((size_t)b << 12);    // b*NDIM
    float* __restrict__ ob = out + ((size_t)b << 24);            // b*NDIM*NDIM

    // --- A fragments for K=0..31 (a0) and K=32..63 (a1) ---
    const _Float16* rowA = xb + ((size_t)(i0 + m) << 6);
    v16h a0, a1;
    {
        v8h lo0 = *(const v8h*)(rowA +      16 * g);
        v8h hi0 = *(const v8h*)(rowA + 8 +  16 * g);
        v8h lo1 = *(const v8h*)(rowA + 32 + 16 * g);
        v8h hi1 = *(const v8h*)(rowA + 40 + 16 * g);
#pragma unroll
        for (int k = 0; k < 8; ++k) {
            a0[k] = lo0[k]; a0[k + 8] = hi0[k];
            a1[k] = lo1[k]; a1[k + 8] = hi1[k];
        }
    }

    // --- row-norm reciprocals for rows i0+8g .. i0+8g+7 (matches C layout) ---
    v4f rni0 = *(const v4f*)(rb + i0 + 8 * g);
    v4f rni1 = *(const v4f*)(rb + i0 + 8 * g + 4);
    float rni[8] = { rni0[0], rni0[1], rni0[2], rni0[3],
                     rni1[0], rni1[1], rni1[2], rni1[3] };

#pragma unroll
    for (int t = 0; t < 4; ++t) {
        const int j0 = j0base + (t << 4);
        const _Float16* rowB = xb + ((size_t)(j0 + m) << 6);
        v16h b0 = *(const v16h*)(rowB +      16 * g);   // K = 16g..16g+15
        v16h b1 = *(const v16h*)(rowB + 32 + 16 * g);   // K = 32+16g..

        v8f acc = {};
        acc = __builtin_amdgcn_wmma_f32_16x16x32_f16(
            /*neg_a=*/false, a0, /*neg_b=*/false, b0,
            /*c_mod=*/(short)0, acc, /*reuse_a=*/false, /*reuse_b=*/false);
        acc = __builtin_amdgcn_wmma_f32_16x16x32_f16(
            false, a1, false, b1, (short)0, acc, false, false);

        const float rnj = rb[j0 + m];
        float* ocol = ob + ((size_t)(i0 + 8 * g) << 12) + (j0 + m);
#pragma unroll
        for (int r = 0; r < 8; ++r) {
            ocol[(size_t)r << 12] = acc[r] * (rni[r] * rnj);
        }
    }
}

// ---------------------------------------------------------------------------
// Harness entry
// ---------------------------------------------------------------------------
extern "C" void kernel_launch(void* const* d_in, const int* in_sizes, int n_in,
                              void* d_out, int out_size, void* d_ws, size_t ws_size,
                              hipStream_t stream) {
    const float* x = (const float*)d_in[0];
    float* out = (float*)d_out;

    // Workspace layout: [0, 2MB) f16 Xt ; [2MB, 2MB+64KB) f32 rnorm
    _Float16* xt = (_Float16*)d_ws;
    float* rnorm = (float*)((char*)d_ws + (size_t)BATCH * NDIM * CDIM * sizeof(_Float16));

    // Phase 1: BATCH*NDIM = 16384 threads
    prep_kernel<<<dim3((BATCH * NDIM) / 256), dim3(256), 0, stream>>>(x, xt, rnorm);

    // Phase 2: z = batch, y = 256 i-tiles, x = 8 blocks of (8 waves x 64 cols)
    gram_kernel<<<dim3(NDIM / 512, NDIM / 16, BATCH), dim3(256), 0, stream>>>(xt, rnorm, out);
}
